// SecondStageModelV2_29935922053200
// MI455X (gfx1250) — matrix operand order repaired and checked by reading, not compiled, not run on hardware
//
#include <hip/hip_runtime.h>
#include <hip/hip_bf16.h>
#include <stdint.h>

// ---------------------------------------------------------------------------
// Problem constants (from reference)
// ---------------------------------------------------------------------------
#define BB     8
#define PP     25
#define NCROP  (BB * PP)          // 200
#define DD     32                 // image depth
#define HH     320
#define WW     320
#define CROP   80
#define DEPTH  6
#define XYMAX  80

// conv1: in (1ch eff, 6,80,80) -> out (32ch, 3,40,40), k3 s2 SAME (pad lo=0)
#define C1     32
#define Z1     3
#define Y1     40
#define X1     40
// conv2: in (32ch, 3,40,40) -> out (128ch, 2,20,20), k3 s2 SAME (z lo=1, hw lo=0)
#define C2     128
#define Z2     2
#define Y2     20
#define X2     20
#define NPOS   (Z2 * Y2 * X2)     // 800 = 50 tiles of 16
#define NTILES 50
#define KTAPS  27
#define SPLITS 5                  // position-splits (split-K style partial max)
#define TPB    (NTILES / SPLITS)  // 10 tiles per block

typedef __attribute__((ext_vector_type(16))) _Float16 v16h;
typedef __attribute__((ext_vector_type(8)))  float    v8f;

// ---------------------------------------------------------------------------
// K0: pack conv2 weights into WMMA A-fragment layout (f16) + conv1 summed
//     weights (input channels are a broadcast -> sum over them).
//
// A-fragment (16-bit, 16x32, CDNA5 ISA): lane L (m = L%16, hi = L/16),
// v16h element e -> k = 16*hi + e (contiguous order, same as the B reads in K2).
// Packed layout: w2p[((ct*27+tap)*32+lane)*16 + e]
// ---------------------------------------------------------------------------
__global__ void k0_pack(const float* __restrict__ conv1_w,   // (32,3,3,3,3)
                        const float* __restrict__ conv2_w,   // (128,32,3,3,3)
                        float* __restrict__ w1sum,           // (32,27)
                        _Float16* __restrict__ w2p)          // 8*27*32*16
{
    int tid = blockIdx.x * blockDim.x + threadIdx.x;

    if (tid < C1 * KTAPS) {                   // conv1 effective weights
        int o = tid / KTAPS, tap = tid % KTAPS;
        float s = 0.f;
        for (int i = 0; i < 3; ++i)
            s += conv1_w[(o * 3 + i) * KTAPS + tap];
        w1sum[tid] = s;
    }
    if (tid < 8 * KTAPS * 32 * 16) {          // conv2 fragment pack
        int e    = tid & 15;
        int lane = (tid >> 4) & 31;
        int tc   = tid >> 9;
        int tap  = tc % KTAPS;
        int ct   = tc / KTAPS;
        int m    = ct * 16 + (lane & 15);
        int hi   = lane >> 4;
        int j    = e >> 1, hl = e & 1;
        int k    = 16 * hi + e;
        w2p[tid] = (_Float16)conv2_w[(m * 32 + k) * KTAPS + tap];
    }
}

// ---------------------------------------------------------------------------
// K1: fused crop-extract + conv1 (stride 2, SAME) + ReLU -> f16, channel-last
//     h1[n][zo][yo][xo][c], c contiguous (32 f16 = 64B). One thread per
//     spatial output position, computing all 32 channels (27 gathers reused).
//     Whole image tensor (105MB) is L2-resident (192MB L2).
// ---------------------------------------------------------------------------
__global__ void k1_conv1(const float* __restrict__ images,
                         const float* __restrict__ z_pred,
                         const float* __restrict__ xy_pred,
                         const float* __restrict__ w1sum,
                         _Float16* __restrict__ h1)
{
    __shared__ float s_w[C1 * KTAPS];
    for (int i = threadIdx.x; i < C1 * KTAPS; i += blockDim.x)
        s_w[i] = w1sum[i];
    __syncthreads();

    int idx = blockIdx.x * blockDim.x + threadIdx.x;
    if (idx >= NCROP * Z1 * Y1 * X1) return;

    int n   = idx / (Z1 * Y1 * X1);
    int rem = idx % (Z1 * Y1 * X1);
    int zo  = rem / (Y1 * X1);
    int r2  = rem % (Y1 * X1);
    int yo  = r2 / X1;
    int xo  = r2 % X1;
    int b   = n / PP;

    // crop centers (match jnp.round = rint, then clip)
    int zc = (int)rintf(z_pred[n]);
    int xc = (int)rintf((float)WW * xy_pred[n * 2 + 0] / (float)XYMAX);
    int yc = (int)rintf((float)HH * xy_pred[n * 2 + 1] / (float)XYMAX);
    zc = min(max(zc, 0), DD - 1);
    xc = min(max(xc, 0), WW - 1);
    yc = min(max(yc, 0), HH - 1);
    int z0 = zc - DEPTH / 2, y0 = yc - CROP / 2, x0 = xc - CROP / 2;

    float v[KTAPS];
    int t = 0;
    #pragma unroll
    for (int kd = 0; kd < 3; ++kd) {
        int cz = 2 * zo + kd;              // pad lo = 0, cz==6 is OOB pad
        int zi = z0 + cz;
        bool okz = (cz < DEPTH) && ((unsigned)zi < (unsigned)DD);
        #pragma unroll
        for (int kh = 0; kh < 3; ++kh) {
            int cy = 2 * yo + kh;
            int yi = y0 + cy;
            bool oky = okz && (cy < CROP) && ((unsigned)yi < (unsigned)HH);
            #pragma unroll
            for (int kw = 0; kw < 3; ++kw) {
                int cx = 2 * xo + kw;
                int xi = x0 + cx;
                bool ok = oky && (cx < CROP) && ((unsigned)xi < (unsigned)WW);
                v[t++] = ok ? images[(((int64_t)b * DD + zi) * HH + yi) * WW + xi]
                            : 0.0f;
            }
        }
    }

    unsigned pk[16];
    #pragma unroll
    for (int c2 = 0; c2 < 16; ++c2) {
        float a0 = 0.f, a1 = 0.f;
        #pragma unroll
        for (int tt = 0; tt < KTAPS; ++tt) {
            a0 = fmaf(s_w[(2 * c2 + 0) * KTAPS + tt], v[tt], a0);
            a1 = fmaf(s_w[(2 * c2 + 1) * KTAPS + tt], v[tt], a1);
        }
        _Float16 h0  = (_Float16)fmaxf(a0, 0.f);
        _Float16 h1v = (_Float16)fmaxf(a1, 0.f);
        unsigned u0 = (unsigned)__builtin_bit_cast(unsigned short, h0);
        unsigned u1 = (unsigned)__builtin_bit_cast(unsigned short, h1v);
        pk[c2] = u0 | (u1 << 16);
    }
    uint4* dst = (uint4*)(h1 + (int64_t)idx * C1);   // 64B aligned
    dst[0] = make_uint4(pk[0],  pk[1],  pk[2],  pk[3]);
    dst[1] = make_uint4(pk[4],  pk[5],  pk[6],  pk[7]);
    dst[2] = make_uint4(pk[8],  pk[9],  pk[10], pk[11]);
    dst[3] = make_uint4(pk[12], pk[13], pk[14], pk[15]);
}

// ---------------------------------------------------------------------------
// K2: conv2 as implicit GEMM via V_WMMA_F32_16X16X32_F16, fused ReLU +
//     max-pool. Block = 8 waves (wave ct owns output channels [16ct,16ct+16)).
//     blockIdx = (split, crop); each block processes 10 position-tiles.
//
//     Per tile: the 27-tap x 16-pos x 32-ch activation tile (27KB) is staged
//     cooperatively into LDS (zero-fill handles SAME padding once, so the
//     WMMA loop is branch-free), then a fully unrolled 27x WMMA loop runs
//     with linear-stride LDS B reads + linear-stride global A reads.
//     Running ReLU-max kept in registers; one shfl-reduce + plain store at
//     the end -> no atomics, fully deterministic.
//
// B-fragment (dense K=32): lane L holds column nl = L%16, ascending
// K = 16*(L/16) + e -> 16 consecutive channels = one 32B contiguous read.
// ---------------------------------------------------------------------------
__global__ void __launch_bounds__(256)
k2_conv2_wmma(const _Float16* __restrict__ h1,
              const _Float16* __restrict__ w2p,
              float* __restrict__ featp)     // (SPLITS, NCROP, C2) partial max
{
    __shared__ _Float16 sB[KTAPS * 16 * C1];   // 27*16*32 f16 = 27648 B

    const int split = blockIdx.x;          // 0..4
    const int n     = blockIdx.y;          // crop 0..199
    const int tid   = threadIdx.x;
    const int ct    = tid >> 5;            // channel tile 0..7
    const int lane  = tid & 31;
    const int nl    = lane & 15;
    const int hi    = lane >> 4;

    const _Float16* aw = w2p + ((size_t)ct * KTAPS * 32 + lane) * 16;
    const _Float16* bw = sB  + (size_t)nl * C1 + hi * 16;

    float rmax[8];
    #pragma unroll
    for (int r = 0; r < 8; ++r) rmax[r] = 0.0f;   // ReLU floor

    for (int nt0 = 0; nt0 < TPB; ++nt0) {
        const int ntile = split * TPB + nt0;
        __syncthreads();                    // previous tile fully consumed

        // ---- stage B tile: (tap, pos16) pairs, 64B each ----
        for (int pair = tid; pair < KTAPS * 16; pair += 256) {
            const int tap = pair >> 4;
            const int p16 = pair & 15;
            const int kd = tap / 9, kh = (tap / 3) % 3, kw = tap % 3;
            const int pos = ntile * 16 + p16;
            const int zo = pos / (Y2 * X2);
            const int rp = pos % (Y2 * X2);
            const int yo = rp / X2;
            const int xo = rp % X2;
            const int z1 = 2 * zo + kd - 1;            // z pad lo = 1
            const int y1 = 2 * yo + kh;                // pad lo = 0
            const int x1 = 2 * xo + kw;
            uint4* dst = (uint4*)&sB[(size_t)pair * C1];
            if (((unsigned)z1 < (unsigned)Z1) &&
                ((unsigned)y1 < (unsigned)Y1) &&
                ((unsigned)x1 < (unsigned)X1)) {
                const uint4* src = (const uint4*)
                    (h1 + ((((size_t)n * Z1 + z1) * Y1 + y1) * X1 + x1) * C1);
                dst[0] = src[0]; dst[1] = src[1];
                dst[2] = src[2]; dst[3] = src[3];
            } else {
                const uint4 zz = make_uint4(0u, 0u, 0u, 0u);
                dst[0] = zz; dst[1] = zz; dst[2] = zz; dst[3] = zz;
            }
        }
        __syncthreads();

        // ---- branch-free fully unrolled GEMM over 27 taps ----
        v8f acc = {};
        #pragma unroll
        for (int tap = 0; tap < KTAPS; ++tap) {
            v16h bfrag = *(const v16h*)(bw + (size_t)tap * 16 * C1);
            v16h afrag = *(const v16h*)(aw + (size_t)tap * 32 * 16);
            acc = __builtin_amdgcn_wmma_f32_16x16x32_f16(
                      false, afrag, false, bfrag, (short)0, acc, false, false);
        }

        union { v8f v; float f[8]; } u;
        u.v = acc;
        #pragma unroll
        for (int r = 0; r < 8; ++r) rmax[r] = fmaxf(rmax[r], u.f[r]);
    }

    // ---- reduce max over the 16 positions in each 16-lane half ----
    // C/D layout: VGPR r holds m = r (lanes 0-15) and m = r+8 (lanes 16-31).
    #pragma unroll
    for (int r = 0; r < 8; ++r) {
        float vm = rmax[r];
        #pragma unroll
        for (int off = 1; off < 16; off <<= 1)
            vm = fmaxf(vm, __shfl_xor(vm, off, 32));
        if (nl == 0) {
            const int m = ct * 16 + r + 8 * hi;
            featp[((size_t)split * NCROP + n) * C2 + m] = vm;  // unique writer
        }
    }
}

// ---------------------------------------------------------------------------
// K3: fold split partial maxima, LayerNorm + head + ordinal link + softmax.
// ---------------------------------------------------------------------------
__global__ void k3_head(const float* __restrict__ featp,
                        const float* __restrict__ ln_g,
                        const float* __restrict__ ln_b,
                        const float* __restrict__ head_w,
                        const float* __restrict__ head_b,
                        const float* __restrict__ cutpoints,
                        const int*   __restrict__ grade,
                        float* __restrict__ probs_out,   // d_out + 1
                        float* __restrict__ part_wnll,
                        float* __restrict__ part_wt)
{
    const int crop = blockIdx.x;
    const int tid  = threadIdx.x;       // 128
    __shared__ float red[C2];

    float f = 0.0f;
    #pragma unroll
    for (int s = 0; s < SPLITS; ++s)
        f = fmaxf(f, featp[((size_t)s * NCROP + crop) * C2 + tid]);

    red[tid] = f; __syncthreads();
    for (int s = 64; s > 0; s >>= 1) {
        if (tid < s) red[tid] += red[tid + s];
        __syncthreads();
    }
    float mu = red[0] / (float)C2; __syncthreads();

    float d = f - mu;
    red[tid] = d * d; __syncthreads();
    for (int s = 64; s > 0; s >>= 1) {
        if (tid < s) red[tid] += red[tid + s];
        __syncthreads();
    }
    float var = red[0] / (float)C2; __syncthreads();

    float nf = d * rsqrtf(var + 1e-5f) * ln_g[tid] + ln_b[tid];
    red[tid] = nf * head_w[tid]; __syncthreads();
    for (int s = 64; s > 0; s >>= 1) {
        if (tid < s) red[tid] += red[tid + s];
        __syncthreads();
    }

    if (tid == 0) {
        float xp = red[0] + head_b[0];
        float s0 = 1.f / (1.f + expf(-(cutpoints[0] - xp)));
        float s1 = 1.f / (1.f + expf(-(cutpoints[1] - xp)));
        float l0 = s0, l1 = s1 - s0, l2 = 1.f - s1;
        float mx = fmaxf(l0, fmaxf(l1, l2));
        float e0 = expf(l0 - mx), e1 = expf(l1 - mx), e2 = expf(l2 - mx);
        float inv = 1.f / (e0 + e1 + e2);
        float p0 = e0 * inv, p1 = e1 * inv, p2 = e2 * inv;
        probs_out[crop * 3 + 0] = p0;
        probs_out[crop * 3 + 1] = p1;
        probs_out[crop * 3 + 2] = p2;

        int t = grade[crop];
        float p  = (t == 0) ? p0 : ((t == 1) ? p1 : p2);
        p = fminf(fmaxf(p, 1e-5f), 1.f - 1e-5f);
        float wt = (t == 0) ? 1.f : ((t == 1) ? 2.f : 4.f);
        part_wnll[crop] = -wt * logf(p);
        part_wt[crop]   = wt;
    }
}

// K4: deterministic serial loss reduction (200 terms).
__global__ void k4_loss(const float* __restrict__ part_wnll,
                        const float* __restrict__ part_wt,
                        float* __restrict__ out)
{
    if (blockIdx.x == 0 && threadIdx.x == 0) {
        float a = 0.f, w = 0.f;
        for (int i = 0; i < NCROP; ++i) { a += part_wnll[i]; w += part_wt[i]; }
        out[0] = a / w;
    }
}

// ---------------------------------------------------------------------------
// Launch
// ---------------------------------------------------------------------------
static inline size_t alignup(size_t x) { return (x + 255) & ~(size_t)255; }

extern "C" void kernel_launch(void* const* d_in, const int* in_sizes, int n_in,
                              void* d_out, int out_size, void* d_ws, size_t ws_size,
                              hipStream_t stream)
{
    const float* images    = (const float*)d_in[0];
    const float* z_pred    = (const float*)d_in[1];
    const float* xy_pred   = (const float*)d_in[2];
    const int*   grade     = (const int*)  d_in[3];
    const float* conv1_w   = (const float*)d_in[4];
    const float* conv2_w   = (const float*)d_in[5];
    const float* ln_g      = (const float*)d_in[6];
    const float* ln_b      = (const float*)d_in[7];
    const float* head_w    = (const float*)d_in[8];
    const float* head_b    = (const float*)d_in[9];
    const float* cutpoints = (const float*)d_in[10];
    float* out = (float*)d_out;

    // workspace carve
    uint8_t* p = (uint8_t*)d_ws;
    float*    w1sum = (float*)p;    p += alignup(C1 * KTAPS * sizeof(float));
    _Float16* w2p   = (_Float16*)p; p += alignup((size_t)8 * KTAPS * 32 * 16 * sizeof(_Float16));
    float*    featp = (float*)p;    p += alignup((size_t)SPLITS * NCROP * C2 * sizeof(float));
    float*    pwn   = (float*)p;    p += alignup(NCROP * sizeof(float));
    float*    pwt   = (float*)p;    p += alignup(NCROP * sizeof(float));
    _Float16* h1    = (_Float16*)p; // 200*3*40*40*32 f16 = 61.4 MB

    // K0: weight prep (110592 items dominates)
    {
        int nthr = 8 * KTAPS * 32 * 16;
        k0_pack<<<(nthr + 255) / 256, 256, 0, stream>>>(conv1_w, conv2_w,
                                                        w1sum, w2p);
    }
    // K1: crop + conv1 + ReLU -> h1 (f16, channel-last)
    {
        int nthr = NCROP * Z1 * Y1 * X1;   // 960000
        k1_conv1<<<(nthr + 255) / 256, 256, 0, stream>>>(images, z_pred,
                                                         xy_pred, w1sum, h1);
    }
    // K2: WMMA conv2 + ReLU + max-pool (partial over splits)
    {
        dim3 grid(SPLITS, NCROP);
        k2_conv2_wmma<<<grid, 256, 0, stream>>>(h1, w2p, featp);
    }
    // K3: head
    k3_head<<<NCROP, C2, 0, stream>>>(featp, ln_g, ln_b, head_w, head_b,
                                      cutpoints, grade, out + 1, pwn, pwt);
    // K4: loss
    k4_loss<<<1, 32, 0, stream>>>(pwn, pwt, out);

    (void)in_sizes; (void)n_in; (void)out_size; (void)ws_size;
}
